// LSTMsolo_28166395527691
// MI455X (gfx1250) — matrix operand order repaired and checked by reading, not compiled, not run on hardware
//
#include <hip/hip_runtime.h>
#include <math.h>

#define B_ 64
#define T_ 1024
#define D_ 128
#define H_ 512
#define G4_ 2048
#define C_ 2
#define KTOT 640          // D_ + H_ : fused [x_t | h] contraction length
#define NWG 32            // persistent workgroups; each owns 16 hidden cols
#define NCOL 16
#define THREADS 256       // 8 wave32s
#define WUPITCH 648       // padded LDS row pitch (bf16 elems): 1296 B, 16B-aligned,
                          // 324 dwords % 64 banks = 4 -> b128 cols stripe all banks

typedef __attribute__((ext_vector_type(16))) __bf16        v16bf;
typedef __attribute__((ext_vector_type(8)))  float         v8f;
typedef __attribute__((ext_vector_type(4)))  unsigned int  v4u;
typedef __attribute__((ext_vector_type(8)))  unsigned int  v8u;
typedef __attribute__((ext_vector_type(4)))  int           v4i;

#if defined(__has_builtin)
#if __has_builtin(__builtin_amdgcn_global_load_async_to_lds_b128)
#define HAVE_ASYNC_LDS 1
#endif
#if __has_builtin(__builtin_amdgcn_s_wait_asynccnt)
#define HAVE_WAIT_ASYNC 1
#endif
#endif

static __device__ __forceinline__ unsigned short f2bf(float f) {
  unsigned u = __builtin_bit_cast(unsigned, f);
  u += 0x7fffu + ((u >> 16) & 1u);           // round-to-nearest-even
  return (unsigned short)(u >> 16);
}
static __device__ __forceinline__ v16bf make_frag(v4u c0, v4u c1) {
  v8u r;
  r[0] = c0[0]; r[1] = c0[1]; r[2] = c0[2]; r[3] = c0[3];
  r[4] = c1[0]; r[5] = c1[1]; r[6] = c1[2]; r[7] = c1[3];
  return __builtin_bit_cast(v16bf, r);
}
static __device__ __forceinline__ float sigmoidf_(float x) {
  return 1.0f / (1.0f + __expf(-x));         // branch-free, v_exp + v_rcp
}
static __device__ __forceinline__ float tanhf_(float x) {
  // branch-free tanh: 2*sigmoid(2x)-1; saturates correctly at +/-inf
  return __builtin_fmaf(2.0f, sigmoidf_(2.0f * x), -1.0f);
}

// ---- fp32 -> bf16 conversion (for x) --------------------------------------
__global__ void lstm_cvt_bf16(const float* __restrict__ src,
                              unsigned short* __restrict__ dst, int n) {
  int i = blockIdx.x * blockDim.x + threadIdx.x;
  int stride = gridDim.x * blockDim.x;
  for (; i < n; i += stride) dst[i] = f2bf(src[i]);
}

// ---- pack WUt[col][k] = bf16( k<128 ? W[k,col] : U[k-128,col] ) -----------
__global__ void lstm_pack_wut(const float* __restrict__ W,
                              const float* __restrict__ U,
                              unsigned short* __restrict__ WUt) {
  int i = blockIdx.x * blockDim.x + threadIdx.x;
  int stride = gridDim.x * blockDim.x;
  const int n = G4_ * KTOT;
  for (; i < n; i += stride) {
    int col = i / KTOT;
    int k = i - col * KTOT;
    float v = (k < D_) ? W[k * G4_ + col] : U[(k - D_) * G4_ + col];
    WUt[i] = f2bf(v);
  }
}

__global__ void lstm_zero(unsigned int* __restrict__ p, int nwords) {
  int i = blockIdx.x * blockDim.x + threadIdx.x;
  int stride = gridDim.x * blockDim.x;
  for (; i < nwords; i += stride) p[i] = 0u;
}

// ---- persistent recurrent kernel ------------------------------------------
// Workgroup wg owns hidden columns [wg*16, wg*16+16) of all 4 gates.
// Weights slab lives in LDS; per step only x_t / h come from L2.
__global__ void __launch_bounds__(THREADS, 1)
lstm_persistent(const unsigned short* __restrict__ xbf,   // [B,T,D] bf16
                const unsigned short* __restrict__ WUt,   // [2048][640] bf16
                const float* __restrict__ bias,           // [4H]
                unsigned short* __restrict__ hbf0,        // [B,H] bf16
                unsigned short* __restrict__ hbf1,        // [B,H] bf16
                float* __restrict__ hf32,                 // [B,H] f32 (last h)
                float* __restrict__ cf32,                 // [B,H] f32 cell
                unsigned int* __restrict__ barrier_mem) { // [count, gen]
  __shared__ __align__(16) unsigned short wu[64 * WUPITCH]; // ~83 KB weights
  __shared__ float zs[64 * 64];                             // 16 KB gates

  const int wg    = blockIdx.x;       // 0..31
  const int tid   = threadIdx.x;
  const int wave  = tid >> 5;
  const int lane  = tid & 31;
  const int lrow  = lane & 15;        // A-row / B-col within tile
  const int khalf = lane >> 4;        // K-half selector (contiguous k = 16*khalf + s)
  const int mtile = wave & 3;         // 64 batch rows -> 4 M-tiles
  const int gate0 = (wave >> 2) << 1; // this wave computes gates gate0,gate0+1

  unsigned int* bcount = barrier_mem;
  unsigned int* bgen   = barrier_mem + 1;

  // ---- one-time LDS fill of this WG's weight slab (async global->LDS) ----
  // local row r (0..63): gate = r>>4, col = r&15 ; global row in WUt.
  for (int c = tid; c < 64 * (KTOT / 8); c += THREADS) {
    int r  = c / (KTOT / 8);
    int kc = (c - r * (KTOT / 8)) * 8;
    int grow = (r >> 4) * H_ + wg * NCOL + (r & 15);
    const unsigned short* gp = WUt + (size_t)grow * KTOT + kc;
    unsigned short*       lp = &wu[r * WUPITCH + kc];
#ifdef HAVE_ASYNC_LDS
    __builtin_amdgcn_global_load_async_to_lds_b128((v4i*)gp, (v4i*)lp, 0, 0);
#else
    *(v4u*)lp = *(const v4u*)gp;
#endif
  }
#ifdef HAVE_ASYNC_LDS
#ifdef HAVE_WAIT_ASYNC
  __builtin_amdgcn_s_wait_asynccnt(0);
#else
  asm volatile("s_wait_asynccnt 0" ::: "memory");
#endif
#endif
  __syncthreads();

  const int m = mtile * 16 + lrow;    // batch row this lane feeds for A
  // B operand rows in LDS: local row = gate*16 + (lane&15)
  const unsigned short* brow0 = &wu[(gate0 * 16 + lrow) * WUPITCH];
  const unsigned short* brow1 = &wu[((gate0 + 1) * 16 + lrow) * WUPITCH];

  for (int t = 0; t < T_; ++t) {
    const unsigned short* hcur = (t & 1) ? hbf1 : hbf0;
    unsigned short*       hnxt = (t & 1) ? hbf0 : hbf1;

    const unsigned short* xrow = xbf + ((size_t)m * T_ + t) * D_;
    const unsigned short* hrow = hcur + (size_t)m * H_ - D_; // k-128 indexing
    if (t + 1 < T_) __builtin_prefetch((const void*)(xrow + D_), 0, 3);

    v8f acc0 = {0.f, 0.f, 0.f, 0.f, 0.f, 0.f, 0.f, 0.f};
    v8f acc1 = {0.f, 0.f, 0.f, 0.f, 0.f, 0.f, 0.f, 0.f};

#pragma unroll 4
    for (int kk = 0; kk < KTOT / 32; ++kk) {
      const int kBase = kk * 32;
      const unsigned short* asrc = (kBase < D_) ? xrow : hrow;
      // A fragment: lanes 0-15 hold K {0..15}, lanes 16-31 {16..31}
      v4u a0 = *(const v4u*)(asrc + kBase + khalf * 16);
      v4u a1 = *(const v4u*)(asrc + kBase + khalf * 16 + 8);
      v16bf afrag = make_frag(a0, a1);
      // B fragments from LDS: 16 contiguous K per half-wave at fixed column
      const int bo = kBase + khalf * 16;
      v16bf bfrag0 = make_frag(*(const v4u*)(brow0 + bo),
                               *(const v4u*)(brow0 + bo + 8));
      v16bf bfrag1 = make_frag(*(const v4u*)(brow1 + bo),
                               *(const v4u*)(brow1 + bo + 8));
      acc0 = __builtin_amdgcn_wmma_f32_16x16x32_bf16(
          false, afrag, false, bfrag0, (short)0, acc0, false, false);
      acc1 = __builtin_amdgcn_wmma_f32_16x16x32_bf16(
          false, afrag, false, bfrag1, (short)0, acc1, false, false);
    }

    // Stage z into LDS: zs[batch][gate*16 + col]
    {
      const int n0 = gate0 * 16 + lrow;
#pragma unroll
      for (int r = 0; r < 8; ++r) {
        int mm = mtile * 16 + r + khalf * 8;   // C/D VGPR layout
        zs[mm * 64 + n0]      = acc0[r];
        zs[mm * 64 + n0 + 16] = acc1[r];
      }
    }
    __syncthreads();

    // Elementwise gate update: 64 batches x 16 cols = 1024 items / 256 threads
#pragma unroll
    for (int j = 0; j < 4; ++j) {
      int idx = tid * 4 + j;
      int bm  = idx >> 4;
      int ci  = idx & 15;
      int col = wg * NCOL + ci;
      float zi = zs[bm * 64 + 0  + ci] + bias[0 * H_ + col];
      float zf = zs[bm * 64 + 16 + ci] + bias[1 * H_ + col];
      float zg = zs[bm * 64 + 32 + ci] + bias[2 * H_ + col];
      float zo = zs[bm * 64 + 48 + ci] + bias[3 * H_ + col];
      float ig = sigmoidf_(zi);
      float fg = sigmoidf_(zf);
      float gg = tanhf_(zg);
      float og = sigmoidf_(zo);
      size_t hidx = (size_t)bm * H_ + col;
      float cNew = fg * cf32[hidx] + ig * gg;
      cf32[hidx] = cNew;
      float hNew = og * tanhf_(cNew);
      hnxt[hidx] = f2bf(hNew);
      hf32[hidx] = hNew;
    }
    __threadfence();   // make h_next visible device-wide before barrier

    // Grid-wide sense-reversal barrier between timesteps
    __syncthreads();
    if (tid == 0) {
      unsigned g = __hip_atomic_load(bgen, __ATOMIC_ACQUIRE,
                                     __HIP_MEMORY_SCOPE_AGENT);
      unsigned arrived = atomicAdd(bcount, 1u);
      if (arrived == NWG - 1) {
        __hip_atomic_store(bcount, 0u, __ATOMIC_RELAXED,
                           __HIP_MEMORY_SCOPE_AGENT);
        __threadfence();
        atomicAdd(bgen, 1u);
      } else {
        while (__hip_atomic_load(bgen, __ATOMIC_ACQUIRE,
                                 __HIP_MEMORY_SCOPE_AGENT) == g) {
          __builtin_amdgcn_s_sleep(2);
        }
      }
    }
    __syncthreads();
    __threadfence();
  }
}

// ---- final FC: out[64,2] = h_last @ Wfc + bfc -----------------------------
__global__ void lstm_fc(const float* __restrict__ hf32,
                        const float* __restrict__ Wfc,
                        const float* __restrict__ bfc,
                        float* __restrict__ out) {
  int tid = threadIdx.x;
  if (tid >= B_ * C_) return;
  int m = tid >> 1, cls = tid & 1;
  float s = bfc[cls];
  for (int k = 0; k < H_; ++k) s += hf32[m * H_ + k] * Wfc[k * C_ + cls];
  out[tid] = s;
}

extern "C" void kernel_launch(void* const* d_in, const int* in_sizes, int n_in,
                              void* d_out, int out_size, void* d_ws,
                              size_t ws_size, hipStream_t stream) {
  const float* x   = (const float*)d_in[0];
  const float* W   = (const float*)d_in[1];
  const float* U   = (const float*)d_in[2];
  const float* b   = (const float*)d_in[3];
  const float* Wfc = (const float*)d_in[4];
  const float* bfc = (const float*)d_in[5];
  float* out = (float*)d_out;

  char* ws = (char*)d_ws;
  size_t off = 0;
  auto alloc = [&](size_t bytes) -> char* {
    char* p = ws + off;
    off = (off + bytes + 255) & ~(size_t)255;
    return p;
  };
  unsigned short* xbf  = (unsigned short*)alloc((size_t)B_ * T_ * D_ * 2);
  unsigned short* WUt  = (unsigned short*)alloc((size_t)G4_ * KTOT * 2);
  unsigned short* hbf0 = (unsigned short*)alloc((size_t)B_ * H_ * 2);
  unsigned short* hbf1 = (unsigned short*)alloc((size_t)B_ * H_ * 2);
  float*          hf32 = (float*)alloc((size_t)B_ * H_ * 4);
  float*          cf32 = (float*)alloc((size_t)B_ * H_ * 4);
  unsigned int*   barrier_mem = (unsigned int*)alloc(256);

  // Zero the state region (hbf0 .. barrier) every call (deterministic).
  size_t zero_bytes = (size_t)((ws + off) - (char*)hbf0);
  int    zero_words = (int)(zero_bytes / 4);

  lstm_cvt_bf16<<<1024, 256, 0, stream>>>(x, xbf, B_ * T_ * D_);
  lstm_pack_wut<<<1024, 256, 0, stream>>>(W, U, WUt);
  lstm_zero<<<256, 256, 0, stream>>>((unsigned int*)hbf0, zero_words);
  lstm_persistent<<<NWG, THREADS, 0, stream>>>(xbf, WUt, b, hbf0, hbf1, hf32,
                                               cf32, barrier_mem);
  lstm_fc<<<1, 128, 0, stream>>>(hf32, Wfc, bfc, out);
}
